// MHSelfAttention_36103495090445
// MI455X (gfx1250) — matrix operand order repaired and checked by reading, not compiled, not run on hardware
//
#include <hip/hip_runtime.h>
#include <cmath>

// ---------------------------------------------------------------------------
// MI455X (gfx1250) Enformer-style MHSA with relative position bias.
// wave32, WMMA bf16 (v_wmma_f32_16x16x32_bf16), fp32 accumulation.
// Attention tiles are staged into LDS by the Tensor Data Mover (TDM,
// tensor_load_to_lds + s_wait_tensorcnt) when the builtin is available.
// ---------------------------------------------------------------------------

typedef __attribute__((ext_vector_type(16))) __bf16 v16bf;
typedef __attribute__((ext_vector_type(8)))  float  v8f;
typedef __attribute__((ext_vector_type(4)))  unsigned int v4u;
typedef __attribute__((ext_vector_type(8)))  int v8i;
typedef __attribute__((ext_vector_type(4)))  int v4i;

#define T_SEQ 1536
#define BATCH 4
#define HEADS 8
#define DKK   64
#define DVV   192
#define CIN   1536
#define LPOS  3071          // 2T-1
#define LPAD  3072          // padded rows for GEMM tiling
#define FEAT  192           // 6*NB
#define HDK   512           // H*DK
#define HDV   1536          // H*DV

#if defined(__gfx1250__) && __has_builtin(__builtin_amdgcn_tensor_load_to_lds)
#define USE_TDM 1
#else
#define USE_TDM 0
#endif

__device__ __forceinline__ v8f wmma_bf16(v16bf a, v16bf b, v8f c) {
  return __builtin_amdgcn_wmma_f32_16x16x32_bf16(false, a, false, b, (short)0, c,
                                                 false, false);
}

// A-fragment for one 32-wide K chunk: base points at the lane's row, k-chunk 0.
__device__ __forceinline__ v16bf load_frag_a(const __bf16* base, int g) {
  union { v16bf v; uint4 u[2]; } t;
  t.u[0] = *(const uint4*)(base + g * 8);
  t.u[1] = *(const uint4*)(base + 16 + g * 8);
  return t.v;
}
// B-fragment for one 32-wide K chunk: base points at the lane's column (K-major).
__device__ __forceinline__ v16bf load_frag_b(const __bf16* base, int g) {
  union { v16bf v; uint4 u[2]; } t;
  t.u[0] = *(const uint4*)(base + g * 8);
  t.u[1] = *(const uint4*)(base + 16 + g * 8);
  return t.v;
}

#if USE_TDM
// Issue a TDM 2-D tile load: tile_d1 rows of tile_d0 bf16 elements, row stride
// stride0 (elements), from global gaddr to LDS byte offset lds_off.
// D# bit layout per CDNA5 ISA ch.8 (group0: count/lds/global/type,
// group1: data_size, tensor dims, tile dims, dim0 stride).
__device__ __forceinline__ void tdm_load_2d(unsigned lds_off, const void* gaddr,
                                            unsigned tensor_d0, unsigned tensor_d1,
                                            unsigned long long stride0,
                                            unsigned tile_d0, unsigned tile_d1) {
  unsigned long long ga = (unsigned long long)(uintptr_t)gaddr;
  v4u g0;
  g0.x = 1u;                                            // count=1, user mode
  g0.y = lds_off;                                       // lds_addr (bytes)
  g0.z = (unsigned)(ga & 0xffffffffu);                  // global_addr[31:0]
  g0.w = (unsigned)((ga >> 32) & 0x01ffffffu) | (2u << 30);  // [56:32] | type=2
  v8i g1;
  g1[0] = (int)(1u << 16);                              // data_size=1 -> 2 bytes
  g1[1] = (int)((tensor_d0 & 0xffffu) << 16);           // tensor_dim0[15:0]
  g1[2] = (int)((tensor_d0 >> 16) | ((tensor_d1 & 0xffffu) << 16));
  g1[3] = (int)((tensor_d1 >> 16) | ((tile_d0 & 0xffffu) << 16));
  g1[4] = (int)(tile_d1 & 0xffffu);                     // tile_dim1 (tile_dim2=0)
  g1[5] = (int)(unsigned)(stride0 & 0xffffffffu);       // dim0 stride [31:0]
  g1[6] = (int)(unsigned)((stride0 >> 32) & 0xffffu);   // dim0 stride [47:32]
  g1[7] = 0;
  v4i z4 = {0, 0, 0, 0};                                // 2-D: groups 2/3 unused
#if __clang_major__ >= 23
  v8i z8 = {0, 0, 0, 0, 0, 0, 0, 0};
  __builtin_amdgcn_tensor_load_to_lds(g0, g1, z4, z4, z8, 0);
#else
  __builtin_amdgcn_tensor_load_to_lds(g0, g1, z4, z4, 0);
#endif
}
#endif

// ---------------------------------------------------------------------------
// fp32 -> bf16 cast
// ---------------------------------------------------------------------------
__global__ __launch_bounds__(256) void cast_f32_bf16(const float* __restrict__ in,
                                                     __bf16* __restrict__ out, int n) {
  int i = (blockIdx.x * 256 + threadIdx.x) * 4;
  if (i + 3 < n) {
    float4 v = *(const float4*)(in + i);
    out[i + 0] = (__bf16)v.x; out[i + 1] = (__bf16)v.y;
    out[i + 2] = (__bf16)v.z; out[i + 3] = (__bf16)v.w;
  } else {
    for (; i < n; ++i) out[i] = (__bf16)in[i];
  }
}

// ---------------------------------------------------------------------------
// Enformer positional basis features -> pe[LPAD][192] bf16
// grid.x = 96: cls = blk/32 (0=exp, 1=central mask, 2=gamma), j = blk%32
// ---------------------------------------------------------------------------
__global__ __launch_bounds__(128) void pos_feats_kernel(__bf16* __restrict__ pe) {
  const float sl = (float)T_SEQ;
  const int cls = blockIdx.x >> 5;
  const int j   = blockIdx.x & 31;
  const int tid = threadIdx.x;
  const int col = cls * 32 + j;
  __shared__ float red[128];

  if (cls == 2) {  // gamma pdf, normalized by max over positions
    const float stddev  = sl / 64.0f;                       // sl/(2*nb)
    const float mean    = 48.0f + (sl - 48.0f) * (float)j / 31.0f;
    const float conc    = (mean / stddev) * (mean / stddev);
    const float rate    = mean / (stddev * stddev);
    const float lognorm = lgammaf(conc) - conc * logf(rate);
    float mx = 0.0f;
    for (int p = tid; p <= T_SEQ - 1; p += 128) {
      float prob = (p == 0) ? 1e-8f
                 : expf((conc - 1.0f) * logf((float)p) - rate * (float)p - lognorm) + 1e-8f;
      mx = fmaxf(mx, prob);
    }
    red[tid] = mx;
    __syncthreads();
    for (int s = 64; s > 0; s >>= 1) {
      if (tid < s) red[tid] = fmaxf(red[tid], red[tid + s]);
      __syncthreads();
    }
    const float pmax = red[0];
    for (int l = tid; l < LPOS; l += 128) {
      float pos  = fabsf((float)(l - (T_SEQ - 1)));
      float prob = (pos == 0.0f) ? 1e-8f
                 : expf((conc - 1.0f) * logf(pos) - rate * pos - lognorm) + 1e-8f;
      float f  = prob / pmax;
      float sg = (l > T_SEQ - 1) ? 1.0f : ((l < T_SEQ - 1) ? -1.0f : 0.0f);
      pe[(size_t)l * FEAT + col]      = (__bf16)f;
      pe[(size_t)l * FEAT + 96 + col] = (__bf16)(sg * f);
    }
  } else if (cls == 0) {  // exponential
    const float maxr = log2f(sl);
    const float hl   = exp2f(3.0f + (maxr - 3.0f) * (float)j / 31.0f);
    for (int l = tid; l < LPOS; l += 128) {
      float pos = fabsf((float)(l - (T_SEQ - 1)));
      float f   = exp2f(-pos / hl);
      float sg  = (l > T_SEQ - 1) ? 1.0f : ((l < T_SEQ - 1) ? -1.0f : 0.0f);
      pe[(size_t)l * FEAT + col]      = (__bf16)f;
      pe[(size_t)l * FEAT + 96 + col] = (__bf16)(sg * f);
    }
  } else {  // central mask
    const float width = exp2f((float)(j + 1)) - 1.0f;
    for (int l = tid; l < LPOS; l += 128) {
      float pos = fabsf((float)(l - (T_SEQ - 1)));
      float f   = (width > pos) ? 1.0f : 0.0f;
      float sg  = (l > T_SEQ - 1) ? 1.0f : ((l < T_SEQ - 1) ? -1.0f : 0.0f);
      pe[(size_t)l * FEAT + col]      = (__bf16)f;
      pe[(size_t)l * FEAT + 96 + col] = (__bf16)(sg * f);
    }
  }
  if (tid == 0) {  // zero the padded row LPOS (also makes rK pad row zero)
    pe[(size_t)LPOS * FEAT + col]      = (__bf16)0.0f;
    pe[(size_t)LPOS * FEAT + 96 + col] = (__bf16)0.0f;
  }
}

// ---------------------------------------------------------------------------
// Generic bf16 WMMA GEMM: C[M,N] = A[M,K] * B[K,N]  (row-major bf16 inputs)
// block = 256 threads (8 waves), tile 128x128, BK=32 (one wmma k-step)
// LDS A tile: [128 rows][32 k]; LDS B tile: [128 cols][32 k] (K-contiguous).
// epilogue modes: 0 = fp32 row-major
//                 1 = bf16 head-split  [B,H,Trows,Dh]
//                 2 = fp32 row-major + bias[n]
//                 3 = bf16 head-split transposed [B,H,Dh,Trows]
// M % 128 == 0, N % 128 == 0, K % 32 == 0
// ---------------------------------------------------------------------------
__global__ __launch_bounds__(256) void gemm_bf16_kernel(
    const __bf16* __restrict__ A, const __bf16* __restrict__ B,
    float* outF, __bf16* outB, const float* __restrict__ bias,
    int M, int N, int K, int mode, int Trows, int Dh, float alpha) {
  __shared__ __bf16 As[128 * 32];
  __shared__ __bf16 Bs[128 * 32];     // transposed: [n][k]
  const int tid = threadIdx.x, lane = tid & 31, wid = tid >> 5;
  const int wm = wid >> 2, wn = wid & 3;       // 2 x 4 wave grid
  const int m0 = blockIdx.y * 128, n0 = blockIdx.x * 128;
  const int g = lane >> 4, ln = lane & 15;

  v8f acc[4][2];
  for (int f = 0; f < 4; ++f)
    for (int n = 0; n < 2; ++n)
      for (int r = 0; r < 8; ++r) acc[f][n][r] = 0.0f;

  const int ar = tid >> 1, ac = (tid & 1) * 16;   // A: 128 rows x 32 k
  const int br = tid >> 3, bc = (tid & 7) * 16;   // B: 32 rows(k) x 128 cols(n)

  for (int k0 = 0; k0 < K; k0 += 32) {
    {
      const uint4* s = (const uint4*)(A + (size_t)(m0 + ar) * K + k0 + ac);
      uint4* d = (uint4*)(&As[ar * 32 + ac]);
      d[0] = s[0]; d[1] = s[1];
    }
    {
      // vectorized global read of 16 n-consecutive halves of row k=br,
      // scatter-store K-major into Bs
      const uint4* s = (const uint4*)(B + (size_t)(k0 + br) * N + n0 + bc);
      union { uint4 u[2]; __bf16 h[16]; } tb;
      tb.u[0] = s[0]; tb.u[1] = s[1];
#pragma unroll
      for (int i = 0; i < 16; ++i) Bs[(bc + i) * 32 + br] = tb.h[i];
    }
    if (k0 + 32 < K) {  // global_prefetch_b8 of next tiles
      __builtin_prefetch(A + (size_t)(m0 + ar) * K + k0 + 32 + ac, 0, 0);
      __builtin_prefetch(B + (size_t)(k0 + 32 + br) * N + n0 + bc, 0, 0);
    }
    __syncthreads();

    v16bf af[4], bfr[2];
    for (int f = 0; f < 4; ++f) {
      int row = wm * 64 + f * 16 + ln;
      af[f] = load_frag_a(&As[row * 32], g);
    }
    for (int n = 0; n < 2; ++n) {
      int col = wn * 32 + n * 16 + ln;
      bfr[n] = load_frag_b(&Bs[col * 32], g);
    }
    for (int f = 0; f < 4; ++f)
      for (int n = 0; n < 2; ++n)
        acc[f][n] = wmma_bf16(af[f], bfr[n], acc[f][n]);
    __syncthreads();
  }

  for (int f = 0; f < 4; ++f)
    for (int n = 0; n < 2; ++n)
      for (int r = 0; r < 8; ++r) {
        int ml = m0 + wm * 64 + f * 16 + r + g * 8;  // C/D layout: M = r | r+8
        int nl = n0 + wn * 32 + n * 16 + ln;
        float v = acc[f][n][r] * alpha;
        if (mode == 0) {
          outF[(size_t)ml * N + nl] = v;
        } else if (mode == 1) {
          int bb = ml / Trows, tt = ml % Trows;
          int hh = nl / Dh,   dd = nl % Dh;
          outB[(((size_t)bb * HEADS + hh) * Trows + tt) * Dh + dd] = (__bf16)v;
        } else if (mode == 2) {
          outF[(size_t)ml * N + nl] = v + bias[nl];
        } else {  // mode 3: head-split transposed [B,H,Dh,Trows]
          int bb = ml / Trows, tt = ml % Trows;
          int hh = nl / Dh,   dd = nl % Dh;
          outB[(((size_t)bb * HEADS + hh) * Dh + dd) * Trows + tt] = (__bf16)v;
        }
      }
}

// ---------------------------------------------------------------------------
// Build Qc = bf16(Q*scale + r_w_bias), Qr = bf16(Q*scale + r_r_bias),
// head-split [B,H,T,DK], from raw fp32 Q [B*T, H*DK]
// ---------------------------------------------------------------------------
__global__ __launch_bounds__(256) void qkv_prep_kernel(
    const float* __restrict__ qraw, const float* __restrict__ rwb,
    const float* __restrict__ rrb, __bf16* __restrict__ qc, __bf16* __restrict__ qr) {
  int idx = blockIdx.x * 256 + threadIdx.x;   // over (B*T)*HDK
  int m = idx >> 9, nn = idx & 511;
  float v = qraw[idx] * 0.125f;               // DK^-0.5 = 64^-0.5
  int b = m / T_SEQ, t = m % T_SEQ, h = nn >> 6, d = nn & 63;
  size_t o = (((size_t)b * HEADS + h) * T_SEQ + t) * DKK + d;
  qc[o] = (__bf16)(v + rwb[nn]);
  qr[o] = (__bf16)(v + rrb[nn]);
}

// ---------------------------------------------------------------------------
// Fused flash attention with relative-shift bias.
// grid = (T/64, B*H); block = 128 (4 waves), wave w owns q rows [q0+16w, +16).
// logits[q,k] = Qc[q].K[k] + Qr[q].rK[T-1+k-q]  (the TXL shift, in-band)
// V comes in head-transposed [B,H,DV,T] so V fragments are K-contiguous.
// K/V/rK-band tiles are staged to LDS by the TDM (wave 0 issues, waits
// TENSORcnt, then the workgroup barrier publishes the tiles).
// ---------------------------------------------------------------------------
__global__ __launch_bounds__(128) void attn_kernel(
    const __bf16* __restrict__ qc, const __bf16* __restrict__ qr,
    const __bf16* __restrict__ kk, const __bf16* __restrict__ vvT,
    const __bf16* __restrict__ rk, __bf16* __restrict__ outb) {
  __shared__ __bf16 Ks[64 * 64];        // key tile   [token][d]   8 KB
  __shared__ __bf16 Vs[192 * 64];       // value tile [dv][token] 24 KB
  __shared__ __bf16 Rs[128 * 64];       // rK band    [j][d]      16 KB
  __shared__ float  Band[4][16 * 32];   // per-wave rel band / P stage  8 KB

  const int tid = threadIdx.x, lane = tid & 31, w = tid >> 5;
  const int g = lane >> 4, ln = lane & 15;
  const int bh = blockIdx.y, b = bh >> 3, h = bh & 7;
  const int q0 = blockIdx.x * 64;
  const int qw0 = q0 + w * 16;
  const size_t headK = (size_t)bh * T_SEQ * DKK;
  const size_t headV = (size_t)bh * DVV * T_SEQ;

  // ---- Q fragments (A layout), both bias flavors, K-chunks 0..31 / 32..63
  v16bf qcF[2], qrF[2];
  {
    const __bf16* qcp = qc + headK + (size_t)(qw0 + ln) * DKK;
    const __bf16* qrp = qr + headK + (size_t)(qw0 + ln) * DKK;
    for (int c = 0; c < 2; ++c) {
      qcF[c] = load_frag_a(qcp + c * 32, g);
      qrF[c] = load_frag_a(qrp + c * 32, g);
    }
  }

  float mrow[8], lrow[8];
  v8f O[12];
  for (int r = 0; r < 8; ++r) { mrow[r] = -1e30f; lrow[r] = 0.0f; }
  for (int nv = 0; nv < 12; ++nv)
    for (int r = 0; r < 8; ++r) O[nv][r] = 0.0f;

  for (int kt = 0; kt < T_SEQ / 64; ++kt) {
    const int k0 = kt * 64;
    const int jb0 = (T_SEQ - 1) + k0 - q0 - 63;   // band start, in [0, LPOS-127+...]
    // ---- LDS staging ---------------------------------------------------
#if USE_TDM
    if (w == 0) {
      // K tile: 64 rows x 64 elems, row stride 64
      tdm_load_2d((unsigned)(uintptr_t)(void*)&Ks[0],
                  kk + headK + (size_t)k0 * DKK, 64u, 64u, 64ull, 64u, 64u);
      // V tile: 192 rows(dv) x 64 tokens, row stride T
      tdm_load_2d((unsigned)(uintptr_t)(void*)&Vs[0],
                  vvT + headV + k0, 64u, 192u, (unsigned long long)T_SEQ, 64u, 192u);
      // rK band: 128 rows x 64 elems, row stride 64 (pad row LPOS is zero)
      tdm_load_2d((unsigned)(uintptr_t)(void*)&Rs[0],
                  rk + ((size_t)h * LPAD + jb0) * DKK, 64u, 128u, 64ull, 64u, 128u);
      __builtin_amdgcn_s_wait_tensorcnt(0);
    }
#else
    {  // K tile: 64x64 halves, 32 halves/thread
      int r = tid >> 1, c = (tid & 1) * 32;
      const uint4* s = (const uint4*)(kk + headK + (size_t)(k0 + r) * DKK + c);
      uint4* d = (uint4*)(&Ks[r * 64 + c]);
      d[0] = s[0]; d[1] = s[1]; d[2] = s[2]; d[3] = s[3];
    }
    {  // V tile (transposed source): 192 rows x 64 tokens, 12 uint4/thread
#pragma unroll
      for (int s4 = 0; s4 < 12; ++s4) {
        int f4 = tid + s4 * 128;          // uint4 index, 0..1535
        int dv = f4 >> 3, c8 = (f4 & 7) * 8;
        const uint4* s = (const uint4*)(vvT + headV + (size_t)dv * T_SEQ + k0 + c8);
        *(uint4*)(&Vs[dv * 64 + c8]) = *s;
      }
    }
    {  // rK band rows: j in [jb0, jb0+127], always within padded buffer
      const uint4* s = (const uint4*)(rk + ((size_t)h * LPAD + jb0 + tid) * DKK);
      uint4* d = (uint4*)(&Rs[tid * 64]);
#pragma unroll
      for (int u = 0; u < 8; ++u) d[u] = s[u];
    }
#endif
    __syncthreads();

    // ---- logits tile: content + shifted rel, 4 subtiles of 16 cols ----
    v8f S[4];
    for (int np = 0; np < 4; ++np) {
      for (int r = 0; r < 8; ++r) S[np][r] = 0.0f;
      // content: B[k=d][n=token] = K[token][d] -> K-contiguous in Ks
      {
        const __bf16* colp = &Ks[(np * 16 + ln) * 64];
        S[np] = wmma_bf16(qcF[0], load_frag_b(colp, g), S[np]);
        S[np] = wmma_bf16(qcF[1], load_frag_b(colp + 32, g), S[np]);
      }
      // rel band: rows Rs[jb + n] and Rs[jb+15+n]
      {
        const int jb = 16 * np - 16 * w + 48;
        v8f D1, D2;
        for (int r = 0; r < 8; ++r) { D1[r] = 0.0f; D2[r] = 0.0f; }
        const __bf16* r1 = &Rs[(jb + ln) * 64];
        D1 = wmma_bf16(qrF[0], load_frag_b(r1, g), D1);
        D1 = wmma_bf16(qrF[1], load_frag_b(r1 + 32, g), D1);
        const __bf16* r2 = &Rs[(jb + 15 + ln) * 64];
        D2 = wmma_bf16(qrF[0], load_frag_b(r2, g), D2);
        D2 = wmma_bf16(qrF[1], load_frag_b(r2 + 32, g), D2);
        // stage 16x31 band to wave-private LDS, read back diagonally shifted
        float* band = Band[w];
        for (int r = 0; r < 8; ++r) {
          int qp = r + 8 * g;
          band[qp * 32 + ln] = D1[r];
        }
        for (int r = 0; r < 8; ++r) {
          int qp = r + 8 * g;
          band[qp * 32 + 15 + ln] = D2[r];
        }
        for (int r = 0; r < 8; ++r) {
          int qp = r + 8 * g;
          S[np][r] += band[qp * 32 + (ln - qp + 15)];
        }
      }
    }

    // ---- online softmax (rows = r + 8g, stats uniform across 16-lane half)
    float t8[8];
    for (int r = 0; r < 8; ++r) {
      float m = S[0][r];
      m = fmaxf(m, S[1][r]); m = fmaxf(m, S[2][r]); m = fmaxf(m, S[3][r]);
      t8[r] = m;
    }
    for (int mk = 1; mk < 16; mk <<= 1)
      for (int r = 0; r < 8; ++r) t8[r] = fmaxf(t8[r], __shfl_xor(t8[r], mk, 32));
    float al[8];
    for (int r = 0; r < 8; ++r) {
      float mn = fmaxf(mrow[r], t8[r]);
      al[r] = __expf(mrow[r] - mn);
      mrow[r] = mn;
    }
    float rs[8];
    for (int r = 0; r < 8; ++r) rs[r] = 0.0f;
    for (int np = 0; np < 4; ++np)
      for (int r = 0; r < 8; ++r) {
        float p = __expf(S[np][r] - mrow[r]);
        S[np][r] = p;
        rs[r] += p;
      }
    for (int mk = 1; mk < 16; mk <<= 1)
      for (int r = 0; r < 8; ++r) rs[r] += __shfl_xor(rs[r], mk, 32);
    for (int r = 0; r < 8; ++r) lrow[r] = lrow[r] * al[r] + rs[r];
    for (int nv = 0; nv < 12; ++nv)
      for (int r = 0; r < 8; ++r) O[nv][r] = O[nv][r] * al[r];

    // ---- P (D layout) -> LDS -> A-fragments; then P @ V ---------------
    __bf16* Ps = (__bf16*)Band[w];   // 16x64 bf16 reuse of the 2KB region
    for (int np = 0; np < 4; ++np)
      for (int r = 0; r < 8; ++r) {
        int qp = r + 8 * g;
        Ps[qp * 64 + np * 16 + ln] = (__bf16)S[np][r];
      }
    v16bf PA0 = load_frag_a(&Ps[ln * 64], g);
    v16bf PA1 = load_frag_a(&Ps[ln * 64] + 32, g);
    for (int nv = 0; nv < 12; ++nv) {
      const __bf16* colv = &Vs[(nv * 16 + ln) * 64];
      O[nv] = wmma_bf16(PA0, load_frag_b(colv, g), O[nv]);
      O[nv] = wmma_bf16(PA1, load_frag_b(colv + 32, g), O[nv]);
    }
    __syncthreads();
  }

  // ---- normalize & write bf16 [B, T, H*DV] ------------------------------
  for (int nv = 0; nv < 12; ++nv)
    for (int r = 0; r < 8; ++r) {
      int qp = r + 8 * g;
      int t = qw0 + qp;
      int col = h * DVV + nv * 16 + ln;
      float val = O[nv][r] / lrow[r];
      outb[((size_t)b * T_SEQ + t) * HDV + col] = (__bf16)val;
    }
}

// ---------------------------------------------------------------------------
// Host launcher
// ---------------------------------------------------------------------------
extern "C" void kernel_launch(void* const* d_in, const int* in_sizes, int n_in,
                              void* d_out, int out_size, void* d_ws, size_t ws_size,
                              hipStream_t stream) {
  const float* x     = (const float*)d_in[0];
  const float* W_q   = (const float*)d_in[1];
  const float* W_k   = (const float*)d_in[2];
  const float* W_v   = (const float*)d_in[3];
  const float* W_rel = (const float*)d_in[4];
  const float* W_out = (const float*)d_in[5];
  const float* b_out = (const float*)d_in[6];
  const float* r_w   = (const float*)d_in[7];
  const float* r_r   = (const float*)d_in[8];
  float* out = (float*)d_out;

  const size_t BT = (size_t)BATCH * T_SEQ;   // 6144
  char* ws = (char*)d_ws;
  size_t off = 0;
  auto alloc = [&](size_t bytes) -> char* {
    char* p = ws + off;
    off = (off + bytes + 255) & ~(size_t)255;
    return p;
  };
  __bf16* xb   = (__bf16*)alloc(BT * CIN * 2);
  __bf16* wqb  = (__bf16*)alloc((size_t)CIN * HDK * 2);
  __bf16* wkb  = (__bf16*)alloc((size_t)CIN * HDK * 2);
  __bf16* wvb  = (__bf16*)alloc((size_t)CIN * HDV * 2);
  __bf16* wob  = (__bf16*)alloc((size_t)HDV * HDV * 2);
  __bf16* wrb  = (__bf16*)alloc((size_t)FEAT * HDK * 2);
  __bf16* peb  = (__bf16*)alloc((size_t)LPAD * FEAT * 2);
  __bf16* rkb  = (__bf16*)alloc((size_t)HEADS * LPAD * DKK * 2);
  float*  qraw = (float*)alloc(BT * HDK * 4);
  __bf16* qcb  = (__bf16*)alloc(BT * HDK * 2);
  __bf16* qrb  = (__bf16*)alloc(BT * HDK * 2);
  __bf16* kb   = (__bf16*)alloc(BT * HDK * 2);
  __bf16* vbt  = (__bf16*)alloc(BT * HDV * 2);   // [B,H,DV,T]
  __bf16* attb = (__bf16*)alloc(BT * HDV * 2);

  auto cast = [&](const float* src, __bf16* dst, size_t n) {
    int blocks = (int)((n + 1023) / 1024);
    cast_f32_bf16<<<blocks, 256, 0, stream>>>(src, dst, (int)n);
  };
  cast(x,     xb,  BT * CIN);
  cast(W_q,   wqb, (size_t)CIN * HDK);
  cast(W_k,   wkb, (size_t)CIN * HDK);
  cast(W_v,   wvb, (size_t)CIN * HDV);
  cast(W_out, wob, (size_t)HDV * HDV);
  cast(W_rel, wrb, (size_t)FEAT * HDK);

  // positional features
  pos_feats_kernel<<<96, 128, 0, stream>>>(peb);

  // rK = pe @ W_rel_k -> head-split bf16 [H, LPAD, DK]
  gemm_bf16_kernel<<<dim3(HDK / 128, LPAD / 128), 256, 0, stream>>>(
      peb, wrb, nullptr, rkb, nullptr, LPAD, HDK, FEAT, 1, LPAD, DKK, 1.0f);

  // Q raw fp32 = x @ W_q
  gemm_bf16_kernel<<<dim3(HDK / 128, (int)BT / 128), 256, 0, stream>>>(
      xb, wqb, qraw, nullptr, nullptr, (int)BT, HDK, CIN, 0, T_SEQ, DKK, 1.0f);
  // Qc/Qr with scale + biases
  qkv_prep_kernel<<<(int)(BT * HDK / 256), 256, 0, stream>>>(qraw, r_w, r_r, qcb, qrb);

  // K -> head-split bf16 [B,H,T,DK]; V -> head-split transposed [B,H,DV,T]
  gemm_bf16_kernel<<<dim3(HDK / 128, (int)BT / 128), 256, 0, stream>>>(
      xb, wkb, nullptr, kb, nullptr, (int)BT, HDK, CIN, 1, T_SEQ, DKK, 1.0f);
  gemm_bf16_kernel<<<dim3(HDV / 128, (int)BT / 128), 256, 0, stream>>>(
      xb, wvb, nullptr, vbt, nullptr, (int)BT, HDV, CIN, 3, T_SEQ, DVV, 1.0f);

  // fused attention
  attn_kernel<<<dim3(T_SEQ / 64, BATCH * HEADS), 128, 0, stream>>>(
      qcb, qrb, kb, vbt, rkb, attb);

  // output projection + bias -> fp32 d_out
  gemm_bf16_kernel<<<dim3(HDV / 128, (int)BT / 128), 256, 0, stream>>>(
      attb, wob, out, nullptr, b_out, (int)BT, HDV, HDV, 2, T_SEQ, DVV, 1.0f);

  (void)in_sizes; (void)n_in; (void)out_size; (void)ws_size;
}
